// GCN_20358144983163
// MI455X (gfx1250) — matrix operand order repaired and checked by reading, not compiled, not run on hardware
//
#include <hip/hip_runtime.h>
#include <hip/hip_bf16.h>
#include <math.h>

typedef __attribute__((ext_vector_type(16))) _Float16 v16h;
typedef __attribute__((ext_vector_type(8)))  _Float16 v8h;
typedef __attribute__((ext_vector_type(8)))  float    v8f;
typedef __attribute__((ext_vector_type(4)))  float    v4f;

#define NN 8192
#define F1 16
#define F2 8

// ---------- degree / dinv ----------
__global__ void k_deg_zero(float* dinv) {
    int i = blockIdx.x * blockDim.x + threadIdx.x;
    if (i < NN) dinv[i] = 0.0f;
}

__global__ void k_deg_count(const int* __restrict__ dst, int E, float* dinv) {
    int e = blockIdx.x * blockDim.x + threadIdx.x;
    if (e < E) atomicAdd(&dinv[dst[e]], 1.0f);
}

// dinv = rsqrt(deg+1); also init x1 = (dinv^2) * W1   (self-loop term of conv1)
__global__ void k_deg_fin_init1(float* dinv, const float* __restrict__ W1, float* x1) {
    int i = blockIdx.x * blockDim.x + threadIdx.x;
    if (i >= NN) return;
    float d = rsqrtf(dinv[i] + 1.0f);
    dinv[i] = d;
    float sw = d * d;
#pragma unroll
    for (int k = 0; k < F1; ++k) x1[i * F1 + k] = sw * W1[i * F1 + k];
}

// ---------- conv1 scatter: x1[dst] += norm * W1[src] ----------
__global__ void k_conv1_scatter(const int* __restrict__ src, const int* __restrict__ dst, int E,
                                const float* __restrict__ dinv, const float* __restrict__ W1,
                                float* x1) {
    int idx = blockIdx.x * blockDim.x + threadIdx.x;
    if (idx >= E * F1) return;
    int e = idx >> 4, k = idx & 15;
    int s = src[e], d = dst[e];
    float nrm = dinv[s] * dinv[d];
    atomicAdd(&x1[d * F1 + k], nrm * W1[s * F1 + k]);
}

// relu(x1 + b1) -> pack f16 for WMMA A operand
__global__ void k_conv1_fin(const float* __restrict__ x1, const float* __restrict__ b1,
                            _Float16* x1h) {
    int idx = blockIdx.x * blockDim.x + threadIdx.x;
    if (idx >= NN * F1) return;
    float v = x1[idx] + b1[idx & 15];
    x1h[idx] = (_Float16)fmaxf(v, 0.0f);
}

// Pre-pack the WMMA B operand (W2, 16x8 padded to 32x16) into the exact
// per-lane register layout: W2h[lane*16 + k].
//   column n = lane&15, h = lane>>4; k<8 -> K = 8h+k -> W2[K*8+n] for n<8, else 0
//   k>=8 : K = 16.. pad -> 0
__global__ void k_prep_w2(const float* __restrict__ W2, _Float16* W2h) {
    int idx = blockIdx.x * blockDim.x + threadIdx.x;   // 0..511
    if (idx >= 32 * 16) return;
    int lane = idx >> 4, k = idx & 15;
    int n = lane & 15, kk = (lane >> 4) * 8 + k;
    float w = (n < F2 && k < 8) ? W2[kk * F2 + n] : 0.0f;
    W2h[idx] = (_Float16)w;
}

// ---------- h2 = relu1 @ W2  via V_WMMA_F32_16X16X32_F16 ----------
// One wave per 16-row tile. Branch-free operand setup: A = one b128 load +
// zero K-pad; B = two b128 loads from the pre-packed W2h.
__global__ void k_gemm_wmma(const _Float16* __restrict__ x1h, const _Float16* __restrict__ W2h,
                            float* __restrict__ h2) {
    int wave = (blockIdx.x * blockDim.x + threadIdx.x) >> 5;  // 0..511, exact grid
    int lane = threadIdx.x & 31;
    int hi   = lane >> 4;      // 0: lanes 0-15, 1: lanes 16-31
    int mn   = lane & 15;

    int tile = wave;           // 512 tiles of 16 rows
    int row  = tile * 16 + mn;

    // A (16x32 f16): lanes 0-15 -> K=0..7, lanes 16-31 -> K=8..15 (slots 0-7);
    // slots 8-15 are K>=16 pad = 0. 16B aligned contiguous load.
    v8h alo = *(const v8h*)(x1h + row * F1 + hi * 8);
    v8h z8  = {};
    v16h a  = __builtin_shufflevector(alo, z8, 0, 1, 2, 3, 4, 5, 6, 7,
                                               8, 9, 10, 11, 12, 13, 14, 15);

    // B: pre-packed per-lane, two contiguous 16B loads.
    v8h b0 = *(const v8h*)(W2h + lane * 16);
    v8h b1 = *(const v8h*)(W2h + lane * 16 + 8);
    v16h b = __builtin_shufflevector(b0, b1, 0, 1, 2, 3, 4, 5, 6, 7,
                                             8, 9, 10, 11, 12, 13, 14, 15);

    v8f c = {};
    c = __builtin_amdgcn_wmma_f32_16x16x32_f16(
            /*neg_a=*/false, a, /*neg_b=*/false, b,
            /*c_mod=*/(short)0, c, /*reuse_a=*/false, /*reuse_b=*/false);

    // D (f32 16x16): VGPR r -> M = r + 8*hi, N = mn. Keep only N < 8.
    if (mn < F2) {
#pragma unroll
        for (int r = 0; r < 8; ++r)
            h2[(tile * 16 + r + hi * 8) * F2 + mn] = c[r];
    }
}

// ---------- conv2 ----------
__global__ void k_conv2_init(const float* __restrict__ dinv, const float* __restrict__ h2,
                             float* x2) {
    int idx = blockIdx.x * blockDim.x + threadIdx.x;
    if (idx >= NN * F2) return;
    int n = idx >> 3;
    float d = dinv[n];
    x2[idx] = d * d * h2[idx];
}

__global__ void k_conv2_scatter(const int* __restrict__ src, const int* __restrict__ dst, int E,
                                const float* __restrict__ dinv, const float* __restrict__ h2,
                                float* x2) {
    int idx = blockIdx.x * blockDim.x + threadIdx.x;
    if (idx >= E * F2) return;
    int e = idx >> 3, k = idx & 7;
    int s = src[e], d = dst[e];
    float nrm = dinv[s] * dinv[d];
    atomicAdd(&x2[d * F2 + k], nrm * h2[s * F2 + k]);
}

__global__ void k_conv2_fin(float* x2, const float* __restrict__ b2) {
    int idx = blockIdx.x * blockDim.x + threadIdx.x;
    if (idx >= NN * F2) return;
    x2[idx] = fmaxf(x2[idx] + b2[idx & 7], 0.0f);
}

// ---------- adjacency copy (the real bottleneck: 512 MB of traffic) ----------
__global__ void k_copy_adj(const v4f* __restrict__ src, v4f* __restrict__ dst, long n4) {
    long i = (long)blockIdx.x * blockDim.x + threadIdx.x;
    long stride = (long)gridDim.x * blockDim.x;
    for (; i < n4; i += stride) {
        v4f v = __builtin_nontemporal_load(&src[i]);   // global_load_b128, NT
        __builtin_nontemporal_store(v, &dst[i]);       // global_store_b128, NT
    }
}

// ---------- edge MLP + orientation scatter ----------
__global__ void k_edge(const int* __restrict__ ue, int U, const float* __restrict__ x2,
                       const float* __restrict__ Wfc, const float* __restrict__ bfc,
                       float* out) {
    int e = blockIdx.x * blockDim.x + threadIdx.x;
    if (e >= U) return;
    int u = ue[2 * e], v = ue[2 * e + 1];
    float s = bfc[0];
#pragma unroll
    for (int k = 0; k < F2; ++k) s += x2[u * F2 + k] * Wfc[k];
#pragma unroll
    for (int k = 0; k < F2; ++k) s += x2[v * F2 + k] * Wfc[F2 + k];
    float o = 1.0f / (1.0f + __expf(-s));
    out[(long)u * NN + v] = o;
    out[(long)v * NN + u] = 1.0f - o;
}

extern "C" void kernel_launch(void* const* d_in, const int* in_sizes, int n_in,
                              void* d_out, int out_size, void* d_ws, size_t ws_size,
                              hipStream_t stream) {
    const float* adj = (const float*)d_in[0];
    const float* W1  = (const float*)d_in[1];
    const float* b1  = (const float*)d_in[2];
    const float* W2  = (const float*)d_in[3];
    const float* b2  = (const float*)d_in[4];
    const float* Wfc = (const float*)d_in[5];
    const float* bfc = (const float*)d_in[6];
    const int*   ei  = (const int*)d_in[7];
    const int*   ue  = (const int*)d_in[8];

    const int E = in_sizes[7] / 2;   // edge_index is (2, E)
    const int U = in_sizes[8] / 2;   // undirected_edges is (U, 2)
    const int* src = ei;
    const int* dst = ei + E;

    // workspace layout (floats): dinv[8192] | x1[8192*16] | h2[8192*8] | x2[8192*8]
    //                            | x1h (8192*16 half) | W2h (512 half)
    float*    dinv = (float*)d_ws;
    float*    x1   = dinv + NN;
    float*    h2   = x1 + NN * F1;
    float*    x2   = h2 + NN * F2;
    _Float16* x1h  = (_Float16*)(x2 + NN * F2);
    _Float16* W2h  = x1h + NN * F1;
    float*    out  = (float*)d_out;

    const int B = 256;

    // degrees -> dinv
    k_deg_zero<<<(NN + B - 1) / B, B, 0, stream>>>(dinv);
    k_deg_count<<<(E + B - 1) / B, B, 0, stream>>>(dst, E, dinv);
    k_deg_fin_init1<<<(NN + B - 1) / B, B, 0, stream>>>(dinv, W1, x1);

    // conv1 + relu -> f16; pre-pack B operand
    k_conv1_scatter<<<(E * F1 + B - 1) / B, B, 0, stream>>>(src, dst, E, dinv, W1, x1);
    k_conv1_fin<<<(NN * F1 + B - 1) / B, B, 0, stream>>>(x1, b1, x1h);
    k_prep_w2<<<2, B, 0, stream>>>(W2, W2h);

    // h2 = relu1 @ W2 via WMMA: 512 tiles, 1 wave each -> 64 blocks of 8 waves (exact)
    k_gemm_wmma<<<(NN / 16) * 32 / B, B, 0, stream>>>(x1h, W2h, h2);

    // conv2 + relu
    k_conv2_init<<<(NN * F2 + B - 1) / B, B, 0, stream>>>(dinv, h2, x2);
    k_conv2_scatter<<<(E * F2 + B - 1) / B, B, 0, stream>>>(src, dst, E, dinv, h2, x2);
    k_conv2_fin<<<(NN * F2 + B - 1) / B, B, 0, stream>>>(x2, b2);

    // stream the 256 MB adjacency into d_out (non-temporal b128)
    long n4 = (long)NN * NN / 4;
    k_copy_adj<<<4096, B, 0, stream>>>((const v4f*)adj, (v4f*)out, n4);

    // per-edge sigmoid + orientation scatter (overwrites ~2*U cells)
    k_edge<<<(U + B - 1) / B, B, 0, stream>>>(ue, U, x2, Wfc, bfc, out);
}
